// DSTGAT_11433202942566
// MI455X (gfx1250) — compile-verified, matrix operand order checked by reading, not run on hardware
//
#include <hip/hip_runtime.h>
#include <hip/hip_bf16.h>
#include <math.h>

// ---------------------------------------------------------------------------
// DSTGAT forward for MI455X (gfx1250, wave32).
// Dense GEMMs ([10000,256]x[256,512] GAT projections, [10000,256]x[256,128]
// fc1) run on v_wmma_f32_16x16x32_f16 (f16 in, f32 accum) with the B panel
// staged into LDS via GLOBAL_LOAD_ASYNC_TO_LDS_B128 (ASYNCcnt) and shared by
// the 5 waves of a block. One wave computes a 16x64 output strip (4 accums);
// per K=32 step: 2 global b128 (A) + 8 ds b128 (4 B frags, all hoisted into
// distinct registers) -> 4 back-to-back v_wmma issues, single dscnt wait.
// Everything else is VALU + L2-resident gather/scatter.
// ---------------------------------------------------------------------------

#define NNODES 10000
#define NEDGES 160000
#define TSEQ   8

typedef _Float16 v16h  __attribute__((ext_vector_type(16)));
typedef _Float16 half8 __attribute__((ext_vector_type(8)));
typedef float    v8f   __attribute__((ext_vector_type(8)));

static __device__ __forceinline__ void atomicMaxF32(float* addr, float v) {
    // classic monotonic reinterpretation trick; init value must be -inf
    if (v >= 0.0f) atomicMax((int*)addr, __float_as_int(v));
    else           atomicMin((unsigned int*)addr, (unsigned int)__float_as_int(v));
}

static __device__ __forceinline__ v16h pack16(half8 lo, half8 hi) {
    v16h r;
    #pragma unroll
    for (int i = 0; i < 8; ++i) { r[i] = lo[i]; r[8 + i] = hi[i]; }
    return r;
}

// ---------------- elementwise utilities ----------------
__global__ void k_fill_f32(float* p, float v, int n) {
    int i = blockIdx.x * blockDim.x + threadIdx.x;
    if (i < n) p[i] = v;
}

__global__ void k_f32_to_f16(const float* __restrict__ in, _Float16* __restrict__ out, int n) {
    int i = blockIdx.x * blockDim.x + threadIdx.x;
    if (i < n) out[i] = (_Float16)in[i];
}

// ---------------- 1x1 conv:  y[n,co,t] = b[co] + sum_ci w[co,ci]*x[n,ci,t] ----------------
__global__ void k_conv1x1(const float* __restrict__ x, const float* __restrict__ w,
                          const float* __restrict__ b, float* __restrict__ y,
                          int N, int Cin, int Cout, int T) {
    int idx = blockIdx.x * blockDim.x + threadIdx.x;
    int total = N * Cout * T;
    if (idx >= total) return;
    int t  = idx % T;
    int co = (idx / T) % Cout;
    int n  = idx / (T * Cout);
    const float* xp = x + (size_t)n * Cin * T + t;
    const float* wp = w + (size_t)co * Cin;
    float s = b[co];
    for (int ci = 0; ci < Cin; ++ci) s += wp[ci] * xp[(size_t)ci * T];
    y[idx] = s;
}

// ---------------- gated dilated TCN: tanh(conv1)*sigmoid(conv2), K=3, SAME pad ----------------
__global__ void k_gated_tcn(const float* __restrict__ x,
                            const float* __restrict__ w1, const float* __restrict__ b1,
                            const float* __restrict__ w2, const float* __restrict__ b2,
                            float* __restrict__ y, int N, int Cin, int Cout, int T, int dil) {
    int idx = blockIdx.x * blockDim.x + threadIdx.x;
    int total = N * Cout * T;
    if (idx >= total) return;
    int t  = idx % T;
    int co = (idx / T) % Cout;
    int n  = idx / (T * Cout);
    float s1 = b1[co], s2 = b2[co];
    const float* xp = x + (size_t)n * Cin * T;
    for (int ci = 0; ci < Cin; ++ci) {
        const float* w1p = w1 + ((size_t)co * Cin + ci) * 3;
        const float* w2p = w2 + ((size_t)co * Cin + ci) * 3;
        #pragma unroll
        for (int k = 0; k < 3; ++k) {
            int tt = t + (k - 1) * dil;
            if (tt >= 0 && tt < T) {
                float v = xp[(size_t)ci * T + tt];
                s1 += w1p[k] * v;
                s2 += w2p[k] * v;
            }
        }
    }
    y[idx] = tanhf(s1) * (1.0f / (1.0f + expf(-s2)));
}

// ---------------- training-mode BatchNorm1d: stats per channel over (N,T) ----------------
__global__ void k_bn_stats(const float* __restrict__ x, float* __restrict__ mean,
                           float* __restrict__ istd, int N, int C, int T) {
    int c = blockIdx.x;
    __shared__ float ss[256], sq[256];
    float s = 0.f, q = 0.f;
    for (int n = threadIdx.x; n < N; n += blockDim.x) {
        const float* p = x + ((size_t)n * C + c) * T;
        #pragma unroll
        for (int t = 0; t < TSEQ; ++t) { float v = p[t]; s += v; q += v * v; }
    }
    ss[threadIdx.x] = s; sq[threadIdx.x] = q;
    __syncthreads();
    for (int k = 128; k > 0; k >>= 1) {
        if ((int)threadIdx.x < k) { ss[threadIdx.x] += ss[threadIdx.x + k]; sq[threadIdx.x] += sq[threadIdx.x + k]; }
        __syncthreads();
    }
    if (threadIdx.x == 0) {
        float cnt = (float)N * (float)T;
        float m = ss[0] / cnt;
        float var = sq[0] / cnt - m * m;
        mean[c] = m;
        istd[c] = rsqrtf(var + 1e-5f);
    }
}

__global__ void k_bn_apply(const float* __restrict__ x, float* __restrict__ y,
                           const float* __restrict__ mean, const float* __restrict__ istd,
                           const float* __restrict__ g, const float* __restrict__ b,
                           int N, int C, int T, _Float16* __restrict__ y16) {
    int idx = blockIdx.x * blockDim.x + threadIdx.x;
    if (idx >= N * C * T) return;
    int c = (idx / T) % C;
    float v = g[c] * (x[idx] - mean[c]) * istd[c] + b[c];
    y[idx] = v;
    if (y16) y16[idx] = (_Float16)v;
}

// ---------------- WMMA GEMM: out[M,Nout] = A[M,K](f16) * Bt[Nout,K]^T(f16) (+bias, relu) ----
// Block = 5 waves (160 threads). Block covers 5 M-tiles x one 64-column N-panel.
// B panel (64 x K f16, 32KB max) is staged global->LDS with async-to-LDS copies
// (ASYNCcnt + s_wait_asynccnt + barrier). Per K=32 step each wave hoists the A
// fragment (2 global b128) and all four B fragments (8 ds b128, distinct regs)
// before issuing 4 v_wmma back-to-back.
// Requires M == gridM*5*16, Nout%64==0, K%32==0, K<=256. EXEC all-ones throughout.
#define GEMM_WAVES 5
#define GEMM_KMAX  256

__global__ void k_wmma_gemm_f16(const _Float16* __restrict__ A, const _Float16* __restrict__ Bt,
                                const float* __restrict__ bias, float* __restrict__ out,
                                int K, int Nout, int relu) {
    __shared__ __align__(16) _Float16 Bs[64 * GEMM_KMAX];

    const int npanels = Nout >> 6;
    const int mg = blockIdx.x / npanels;       // M-tile group
    const int np = blockIdx.x % npanels;       // 64-column N panel
    const int lane = threadIdx.x & 31;
    const int wave = threadIdx.x >> 5;

    // ---- stage B panel [64,K] (contiguous rows of Bt) into LDS, async ----
    {
        const _Float16* gpanel = Bt + (size_t)np * 64 * K;
        const int chunks = (64 * K) >> 3;              // 16B chunks
        const unsigned lds0 = (unsigned)(uintptr_t)(&Bs[0]);
        for (int i = threadIdx.x; i < chunks; i += blockDim.x) {
            unsigned long long ga = (unsigned long long)(uintptr_t)(gpanel + (size_t)i * 8);
            unsigned la = lds0 + (unsigned)i * 16u;
            asm volatile("global_load_async_to_lds_b128 %0, %1, off"
                         :: "v"(la), "v"(ga) : "memory");
        }
        asm volatile("s_wait_asynccnt 0x0" ::: "memory");
    }
    __syncthreads();

    const int mtile = mg * GEMM_WAVES + wave;
    const int r  = lane & 15;          // A row / B column within 16-tile
    const int kb = (lane >> 4) << 3;   // lane-group K offset (0 or 8 halves)

    const _Float16* ap = A + ((size_t)(mtile * 16 + r)) * K + kb;
    const _Float16* bbase = Bs + (size_t)r * K + kb;   // + ns*16*K per subtile

    v8f acc[4] = {{}, {}, {}, {}};
    for (int k0 = 0; k0 < K; k0 += 32) {
        // hoist A fragment (global) and all 4 B fragments (LDS, distinct regs)
        half8 a0 = *(const half8*)(ap + k0);
        half8 a1 = *(const half8*)(ap + k0 + 16);
        v16h av = pack16(a0, a1);
        v16h bv[4];
        #pragma unroll
        for (int ns = 0; ns < 4; ++ns) {
            const _Float16* bp = bbase + (size_t)(ns * 16) * K + k0;
            bv[ns] = pack16(*(const half8*)(bp), *(const half8*)(bp + 16));
        }
        #pragma unroll
        for (int ns = 0; ns < 4; ++ns) {
            acc[ns] = __builtin_amdgcn_wmma_f32_16x16x32_f16(false, av, false, bv[ns],
                                                             (short)0, acc[ns], false, false);
        }
    }

    const int mb = (lane >> 4) << 3;   // D rows 0..7 (lanes 0-15) / 8..15 (lanes 16-31)
    #pragma unroll
    for (int ns = 0; ns < 4; ++ns) {
        const int col = np * 64 + ns * 16 + r;
        float bb = bias ? bias[col] : 0.0f;
        #pragma unroll
        for (int i = 0; i < 8; ++i) {
            float v = acc[ns][i] + bb;
            if (relu) v = fmaxf(v, 0.0f);
            out[((size_t)(mtile * 16 + mb + i)) * Nout + col] = v;
        }
    }
}

// ---------------- GAT attention pieces ----------------
__global__ void k_attn_scores(const float* __restrict__ feat, const float* __restrict__ al,
                              const float* __restrict__ ar, float* __restrict__ el,
                              float* __restrict__ er, int N, int H, int D) {
    int idx = blockIdx.x * blockDim.x + threadIdx.x;
    if (idx >= N * H) return;
    int h = idx % H, n = idx / H;
    const float* f = feat + ((size_t)n * H + h) * D;
    const float* lp = al + (size_t)h * D;
    const float* rp = ar + (size_t)h * D;
    float sl = 0.f, sr = 0.f;
    for (int d = 0; d < D; ++d) { float v = f[d]; sl += v * lp[d]; sr += v * rp[d]; }
    el[idx] = sl; er[idx] = sr;
}

__global__ void k_edge_score_max(const float* __restrict__ el, const float* __restrict__ er,
                                 const int* __restrict__ src, const int* __restrict__ dst,
                                 float* __restrict__ e, float* __restrict__ emax, int E, int H) {
    int idx = blockIdx.x * blockDim.x + threadIdx.x;
    if (idx >= E * H) return;
    int h = idx % H, ed = idx / H;
    int s = src[ed], d = dst[ed];
    float v = el[s * H + h] + er[d * H + h];
    v = (v > 0.f) ? v : 0.2f * v;              // leaky_relu(0.2)
    e[idx] = v;
    atomicMaxF32(&emax[d * H + h], v);
}

__global__ void k_edge_exp_sum(float* __restrict__ e, const int* __restrict__ dst,
                               const float* __restrict__ emax, float* __restrict__ denom,
                               int E, int H) {
    int idx = blockIdx.x * blockDim.x + threadIdx.x;
    if (idx >= E * H) return;
    int h = idx % H, ed = idx / H;
    int d = dst[ed];
    float ee = expf(e[idx] - emax[d * H + h]);
    e[idx] = ee;                                // in-place: e -> exp(e - max)
    atomicAdd(&denom[d * H + h], ee);
}

__global__ void k_edge_alpha(float* __restrict__ e, const int* __restrict__ dst,
                             const float* __restrict__ denom, int E, int H) {
    int idx = blockIdx.x * blockDim.x + threadIdx.x;
    if (idx >= E * H) return;
    int h = idx % H, ed = idx / H;
    e[idx] = e[idx] / (denom[dst[ed] * H + h] + 1e-16f);
}

__global__ void k_scatter_msg(const float* __restrict__ feat, const float* __restrict__ alpha,
                              const int* __restrict__ src, const int* __restrict__ dst,
                              float* __restrict__ gout, int E, int H, int D) {
    int idx = blockIdx.x * blockDim.x + threadIdx.x;
    int HD = H * D;
    if (idx >= E * HD) return;
    int j = idx % HD;
    int ed = idx / HD;
    int h = j / D;
    int s = src[ed], d = dst[ed];
    atomicAdd(&gout[(size_t)d * HD + j], feat[(size_t)s * HD + j] * alpha[ed * H + h]);
}

__global__ void k_gat_bias_relu(float* __restrict__ g, const float* __restrict__ bias,
                                int N, int HD) {
    int idx = blockIdx.x * blockDim.x + threadIdx.x;
    if (idx >= N * HD) return;
    g[idx] = fmaxf(g[idx] + bias[idx % HD], 0.0f);
}

__global__ void k_head_mean(const float* __restrict__ gout, const float* __restrict__ bias,
                            float* __restrict__ out, int N, int H, int D) {
    int idx = blockIdx.x * blockDim.x + threadIdx.x;
    if (idx >= N * D) return;
    int d = idx % D, n = idx / D;
    float s = 0.f;
    for (int h = 0; h < H; ++h)
        s += fmaxf(gout[((size_t)n * H + h) * D + d] + bias[h * D + d], 0.0f);
    out[idx] = s / (float)H;
}

__global__ void k_add_residual(const float* __restrict__ g, const float* __restrict__ res,
                               float* __restrict__ out, int n, int relu,
                               _Float16* __restrict__ out16) {
    int i = blockIdx.x * blockDim.x + threadIdx.x;
    if (i >= n) return;
    float v = g[i] + res[i];
    if (relu) v = fmaxf(v, 0.0f);
    out[i] = v;
    if (out16) out16[i] = (_Float16)v;
}

// ---------------- final fc2 + duplicated logits ----------------
__global__ void k_fc2(const float* __restrict__ h1, const float* __restrict__ w,
                      const float* __restrict__ b, float* __restrict__ out, int N, int K) {
    int n = blockIdx.x * blockDim.x + threadIdx.x;
    if (n >= N) return;
    const float* p = h1 + (size_t)n * K;
    float s = b[0];
    for (int k = 0; k < K; ++k) s += p[k] * w[k];
    out[n] = s;          // logits
    out[N + n] = s;      // logits (duplicated, per reference tuple)
}

// ===========================================================================

static inline int nb(int n) { return (n + 255) / 256; }

extern "C" void kernel_launch(void* const* d_in, const int* in_sizes, int n_in,
                              void* d_out, int out_size, void* d_ws, size_t ws_size,
                              hipStream_t stream) {
    (void)in_sizes; (void)n_in; (void)out_size; (void)ws_size;
    const int N = NNODES, E = NEDGES, T = TSEQ;

    // ---- inputs (setup_inputs insertion order, params flattened recursively) ----
    const float* X      = (const float*)d_in[0];
    const float* in_w   = (const float*)d_in[1];
    const float* in_b   = (const float*)d_in[2];
    // layer 0: idx 3..16
    const float* l0_c1w = (const float*)d_in[3];
    const float* l0_c1b = (const float*)d_in[4];
    const float* l0_c2w = (const float*)d_in[5];
    const float* l0_c2b = (const float*)d_in[6];
    const float* l0_resw= (const float*)d_in[7];
    const float* l0_resb= (const float*)d_in[8];
    const float* l0_bn1g= (const float*)d_in[9];
    const float* l0_bn1b= (const float*)d_in[10];
    const float* l0_bn2g= (const float*)d_in[11];
    const float* l0_bn2b= (const float*)d_in[12];
    const float* l0_gatw= (const float*)d_in[13];
    const float* l0_al  = (const float*)d_in[14];
    const float* l0_ar  = (const float*)d_in[15];
    const float* l0_gatb= (const float*)d_in[16];
    // layer 1: idx 17..30
    const float* l1_c1w = (const float*)d_in[17];
    const float* l1_c1b = (const float*)d_in[18];
    const float* l1_c2w = (const float*)d_in[19];
    const float* l1_c2b = (const float*)d_in[20];
    const float* l1_resw= (const float*)d_in[21];
    const float* l1_resb= (const float*)d_in[22];
    const float* l1_bn1g= (const float*)d_in[23];
    const float* l1_bn1b= (const float*)d_in[24];
    const float* l1_bn2g= (const float*)d_in[25];
    const float* l1_bn2b= (const float*)d_in[26];
    const float* l1_gatw= (const float*)d_in[27];
    const float* l1_al  = (const float*)d_in[28];
    const float* l1_ar  = (const float*)d_in[29];
    const float* l1_gatb= (const float*)d_in[30];
    const float* fc1_w  = (const float*)d_in[31];
    const float* fc1_b  = (const float*)d_in[32];
    const float* fc2_w  = (const float*)d_in[33];
    const float* fc2_b  = (const float*)d_in[34];
    const int*   eidx   = (const int*)  d_in[35];
    const int* src = eidx;
    const int* dst = eidx + E;

    float* out = (float*)d_out;
    float* hidden_out = out + 2 * N;   // [N,32,8] f32

    // ---- workspace arena ----
    char* wp = (char*)d_ws;
    auto alloc = [&](size_t bytes) -> void* {
        void* p = (void*)wp;
        wp += (bytes + 255) & ~(size_t)255;
        return p;
    };
    float*     x0    = (float*)alloc((size_t)N * 32 * T * 4);   // after input conv
    float*     res   = (float*)alloc((size_t)N * 64 * T * 4);   // residual (max size)
    float*     tbuf  = (float*)alloc((size_t)N * 32 * T * 4);   // TCN / bn1 output
    _Float16*  tf16  = (_Float16*)alloc((size_t)N * 256 * 2);   // f16 GEMM operand
    float*     feat  = (float*)alloc((size_t)N * 512 * 4);      // GAT features [N,H*D]
    float*     gout  = (float*)alloc((size_t)N * 512 * 4);      // scatter accumulator
    float*     x1    = (float*)alloc((size_t)N * 512 * 4);      // layer0 output [N,64,8]
    float*     gmean = (float*)alloc((size_t)N * 256 * 4);      // layer1 head-mean
    float*     el    = (float*)alloc((size_t)N * 2 * 4);
    float*     er    = (float*)alloc((size_t)N * 2 * 4);
    float*     emax  = (float*)alloc((size_t)N * 2 * 4);
    float*     den   = (float*)alloc((size_t)N * 2 * 4);
    float*     ebuf  = (float*)alloc((size_t)E * 2 * 4);
    float*     h1    = (float*)alloc((size_t)N * 128 * 4);
    _Float16*  hf16  = (_Float16*)alloc((size_t)N * 256 * 2);
    _Float16*  gw0   = (_Float16*)alloc((size_t)512 * 256 * 2);
    _Float16*  gw1   = (_Float16*)alloc((size_t)512 * 256 * 2);
    _Float16*  fw1   = (_Float16*)alloc((size_t)128 * 256 * 2);
    float*     bmean = (float*)alloc(64 * 4);
    float*     bistd = (float*)alloc(64 * 4);

    // ---- weight conversions (tiny) ----
    k_f32_to_f16<<<nb(512 * 256), 256, 0, stream>>>(l0_gatw, gw0, 512 * 256);
    k_f32_to_f16<<<nb(512 * 256), 256, 0, stream>>>(l1_gatw, gw1, 512 * 256);
    k_f32_to_f16<<<nb(128 * 256), 256, 0, stream>>>(fc1_w,   fw1, 128 * 256);

    // ---- input 1x1 conv: [N,16,8] -> [N,32,8] ----
    k_conv1x1<<<nb(N * 32 * T), 256, 0, stream>>>(X, in_w, in_b, x0, N, 16, 32, T);

    // GEMM grids: 125 M-groups (5 waves * 16 rows each = 10000 rows) x N/64 panels
    const int GEMM_TPB = GEMM_WAVES * 32;        // 160 threads
    const int MGROUPS  = NNODES / (GEMM_WAVES * 16);  // 125

    // ================= Layer 0 (concat=true, dil=1, H=2) =================
    k_conv1x1<<<nb(N * 64 * T), 256, 0, stream>>>(x0, l0_resw, l0_resb, res, N, 32, 64, T);
    k_gated_tcn<<<nb(N * 32 * T), 256, 0, stream>>>(x0, l0_c1w, l0_c1b, l0_c2w, l0_c2b,
                                                    tbuf, N, 32, 32, T, 1);
    k_bn_stats<<<32, 256, 0, stream>>>(tbuf, bmean, bistd, N, 32, T);
    k_bn_apply<<<nb(N * 32 * T), 256, 0, stream>>>(tbuf, tbuf, bmean, bistd, l0_bn1g, l0_bn1b,
                                                   N, 32, T, tf16);
    // feat = h @ gat_w^T : [10000,256] x [256,512]
    k_wmma_gemm_f16<<<MGROUPS * (512 / 64), GEMM_TPB, 0, stream>>>(tf16, gw0, nullptr, feat,
                                                                   256, 512, 0);
    k_attn_scores<<<nb(N * 2), 256, 0, stream>>>(feat, l0_al, l0_ar, el, er, N, 2, 256);
    k_fill_f32<<<nb(N * 2), 256, 0, stream>>>(emax, -INFINITY, N * 2);
    k_fill_f32<<<nb(N * 2), 256, 0, stream>>>(den, 0.0f, N * 2);
    k_fill_f32<<<nb(N * 512), 256, 0, stream>>>(gout, 0.0f, N * 512);
    k_edge_score_max<<<nb(E * 2), 256, 0, stream>>>(el, er, src, dst, ebuf, emax, E, 2);
    k_edge_exp_sum<<<nb(E * 2), 256, 0, stream>>>(ebuf, dst, emax, den, E, 2);
    k_edge_alpha<<<nb(E * 2), 256, 0, stream>>>(ebuf, dst, den, E, 2);
    k_scatter_msg<<<nb(E * 512), 256, 0, stream>>>(feat, ebuf, src, dst, gout, E, 2, 256);
    k_gat_bias_relu<<<nb(N * 512), 256, 0, stream>>>(gout, l0_gatb, N, 512);
    // concat -> [N,512] == [N,64,8]; bn2 over 64 channels, then +residual, relu
    k_bn_stats<<<64, 256, 0, stream>>>(gout, bmean, bistd, N, 64, T);
    k_bn_apply<<<nb(N * 512), 256, 0, stream>>>(gout, gout, bmean, bistd, l0_bn2g, l0_bn2b,
                                                N, 64, T, nullptr);
    k_add_residual<<<nb(N * 512), 256, 0, stream>>>(gout, res, x1, N * 512, 1, nullptr);

    // ================= Layer 1 (concat=false, dil=2, H=2, last) =================
    k_conv1x1<<<nb(N * 32 * T), 256, 0, stream>>>(x1, l1_resw, l1_resb, res, N, 64, 32, T);
    k_gated_tcn<<<nb(N * 32 * T), 256, 0, stream>>>(x1, l1_c1w, l1_c1b, l1_c2w, l1_c2b,
                                                    tbuf, N, 64, 32, T, 2);
    k_bn_stats<<<32, 256, 0, stream>>>(tbuf, bmean, bistd, N, 32, T);
    k_bn_apply<<<nb(N * 32 * T), 256, 0, stream>>>(tbuf, tbuf, bmean, bistd, l1_bn1g, l1_bn1b,
                                                   N, 32, T, tf16);
    k_wmma_gemm_f16<<<MGROUPS * (512 / 64), GEMM_TPB, 0, stream>>>(tf16, gw1, nullptr, feat,
                                                                   256, 512, 0);
    k_attn_scores<<<nb(N * 2), 256, 0, stream>>>(feat, l1_al, l1_ar, el, er, N, 2, 256);
    k_fill_f32<<<nb(N * 2), 256, 0, stream>>>(emax, -INFINITY, N * 2);
    k_fill_f32<<<nb(N * 2), 256, 0, stream>>>(den, 0.0f, N * 2);
    k_fill_f32<<<nb(N * 512), 256, 0, stream>>>(gout, 0.0f, N * 512);
    k_edge_score_max<<<nb(E * 2), 256, 0, stream>>>(el, er, src, dst, ebuf, emax, E, 2);
    k_edge_exp_sum<<<nb(E * 2), 256, 0, stream>>>(ebuf, dst, emax, den, E, 2);
    k_edge_alpha<<<nb(E * 2), 256, 0, stream>>>(ebuf, dst, den, E, 2);
    k_scatter_msg<<<nb(E * 512), 256, 0, stream>>>(feat, ebuf, src, dst, gout, E, 2, 256);
    // mean over heads (bias + relu applied per head first) -> [N,256] == [N,32,8]
    k_head_mean<<<nb(N * 256), 256, 0, stream>>>(gout, l1_gatb, gmean, N, 2, 256);
    k_bn_stats<<<32, 256, 0, stream>>>(gmean, bmean, bistd, N, 32, T);
    k_bn_apply<<<nb(N * 256), 256, 0, stream>>>(gmean, gmean, bmean, bistd, l1_bn2g, l1_bn2b,
                                                N, 32, T, nullptr);
    // last layer: no relu; write hidden directly into d_out, emit f16 for fc1
    k_add_residual<<<nb(N * 256), 256, 0, stream>>>(gmean, res, hidden_out, N * 256, 0, hf16);

    // ================= Output MLP =================
    // fc1: [10000,256] x [256,128] + bias, relu
    k_wmma_gemm_f16<<<MGROUPS * (128 / 64), GEMM_TPB, 0, stream>>>(hf16, fw1, fc1_b, h1,
                                                                   256, 128, 1);
    // fc2: 128-dot per node -> logits written to out[0:N] and out[N:2N]
    k_fc2<<<nb(N), 256, 0, stream>>>(h1, fc2_w, fc2_b, out, N, 128);
}
